// HyenaBlock_5033701671552
// MI455X (gfx1250) — matrix operand order repaired and checked by reading, not compiled, not run on hardware
//
#include <hip/hip_runtime.h>
#include <hip/hip_bf16.h>

#define DEV __device__ __forceinline__

typedef __attribute__((ext_vector_type(16))) __bf16 v16bf;
typedef __attribute__((ext_vector_type(8)))  float  v8f;
typedef int v4i __attribute__((vector_size(16)));
typedef __attribute__((address_space(1))) v4i* gas_v4i;
typedef __attribute__((address_space(3))) v4i* las_v4i;

typedef unsigned u32x4 __attribute__((vector_size(16)));
typedef int i32x8 __attribute__((vector_size(32)));
typedef int i32x4 __attribute__((vector_size(16)));

constexpr int NFFT = 16384;

#if __has_builtin(__builtin_amdgcn_global_load_async_to_lds_b128) && \
    __has_builtin(__builtin_amdgcn_s_wait_asynccnt)
#define USE_ASYNC_LDS 1
#endif

#if __has_builtin(__builtin_amdgcn_tensor_load_to_lds) && \
    __has_builtin(__builtin_amdgcn_s_wait_tensorcnt)
#define USE_TDM 1
#if __has_include(<hip/amd_detail/amd_gfx1250_TDM.h>)
#define TDM_6ARG 1
#endif
#endif

// Copy 16 contiguous bytes global->LDS (ASYNCcnt DMA path on CDNA5).
DEV void cp16(const float* g, float* l) {
#ifdef USE_ASYNC_LDS
  __builtin_amdgcn_global_load_async_to_lds_b128((gas_v4i)(g), (las_v4i)(l), 0,
                                                 0);
#else
  l[0] = g[0]; l[1] = g[1]; l[2] = g[2]; l[3] = g[3];
#endif
}
DEV void cp_wait() {
#ifdef USE_ASYNC_LDS
  __builtin_amdgcn_s_wait_asynccnt(0);
#endif
}

#ifdef USE_TDM
// Issue one TDM 2D tile load: global (elements of 4B, row length dim0,
// dim1 rows, row stride stride0) -> contiguous LDS tile tile1 x tile0.
// D# packing per CDNA5 ISA §8.3-8.4. Caller must be a uniform (wave) branch;
// waits TENSORcnt==0 before returning.
DEV void tdm_load_2d(const void* gaddr, unsigned lds_off, unsigned dim0,
                     unsigned dim1, unsigned stride0, unsigned tile0,
                     unsigned tile1) {
  unsigned long long ga = (unsigned long long)gaddr;
  u32x4 g0;
  g0[0] = 1u;                                  // count=1 valid descriptor
  g0[1] = lds_off;                             // lds_addr (bytes)
  g0[2] = (unsigned)(ga & 0xFFFFFFFFull);      // global_addr[31:0]
  g0[3] = (unsigned)((ga >> 32) & 0x01FFFFFFull) | (2u << 30);  // type=2
  i32x8 g1;
  g1[0] = (int)(2u << 16);                     // data_size=2 (4 bytes)
  g1[1] = (int)((dim0 & 0xFFFFu) << 16);       // tensor_dim0[15:0] @48
  g1[2] = (int)((dim0 >> 16) | ((dim1 & 0xFFFFu) << 16));  // dim0 hi | dim1 lo
  g1[3] = (int)((dim1 >> 16) | (tile0 << 16)); // dim1 hi | tile_dim0
  g1[4] = (int)(tile1 & 0xFFFFu);              // tile_dim1 (tile_dim2=0)
  g1[5] = (int)stride0;                        // tensor_dim0_stride[31:0]
  g1[6] = 0;                                   // stride0 hi | dim1_stride lo
  g1[7] = 0;
  i32x4 z4 = {0, 0, 0, 0};
#ifdef TDM_6ARG
  i32x8 z8 = {0, 0, 0, 0, 0, 0, 0, 0};
  __builtin_amdgcn_tensor_load_to_lds(g0, g1, z4, z4, z8, 0);
#else
  __builtin_amdgcn_tensor_load_to_lds(g0, g1, z4, z4, 0);
#endif
  __builtin_amdgcn_s_wait_tensorcnt(0);
}
DEV unsigned lds_addr_of(const void* p) {
  return (unsigned)(unsigned long long)(__attribute__((address_space(3)))
                                        const char*)(p);
}
#endif  // USE_TDM

// Native RTE float->bf16 (v_cvt on gfx1250; 1 VALU op vs 3 for manual round)
DEV __bf16 f2bf(float f) { return (__bf16)f; }

DEV v8f wmma_bf16(v16bf a, v16bf b, v8f c) {
  return __builtin_amdgcn_wmma_f32_16x16x32_bf16(false, a, false, b, (short)0, c,
                                                 false, false);
}

// ---- fragment loaders (ISA 7.12.2 layouts), f32 LDS staging + cvt ----------
DEV v16bf load_a_ldsf(const float* A, int lda, int row0, int kbase, int lane) {
  int r = row0 + (lane & 15);
  int kb = (lane >= 16) ? 8 : 0;
  v16bf a;
#pragma unroll
  for (int e = 0; e < 16; ++e) {
    int k = kbase + kb + e + ((e >= 8) ? 8 : 0);
    a[e] = f2bf(A[r * lda + k]);
  }
  return a;
}
DEV v16bf load_a_ldsf_t(const float* At, int ldt, int row0, int kbase, int lane) {
  int r = row0 + (lane & 15);
  int kb = (lane >= 16) ? 8 : 0;
  v16bf a;
#pragma unroll
  for (int e = 0; e < 16; ++e) {
    int k = kbase + kb + e + ((e >= 8) ? 8 : 0);
    a[e] = f2bf(At[k * ldt + r]);
  }
  return a;
}
DEV v16bf load_a_lds(const __bf16* A, int lda, int row0, int kbase, int lane) {
  int r = row0 + (lane & 15);
  int kb = (lane >= 16) ? 8 : 0;
  v16bf a;
#pragma unroll
  for (int e = 0; e < 16; ++e) {
    int k = kbase + kb + e + ((e >= 8) ? 8 : 0);
    a[e] = A[r * lda + k];
  }
  return a;
}
DEV v16bf load_b_ldsf(const float* Bm, int ldb, int col0, int lane) {
  int c = col0 + (lane & 15);
  int kb = (lane >= 16) ? 8 : 0;
  v16bf b;
#pragma unroll
  for (int e = 0; e < 16; ++e) b[e] = f2bf(Bm[(kb + e + ((e >= 8) ? 8 : 0)) * ldb + c]);
  return b;
}
DEV v16bf load_b_gl(const float* W, int ldw, int kbase, int col0, int lane) {
  int c = col0 + (lane & 15);
  int kb = (lane >= 16) ? 8 : 0;
  v16bf b;
#pragma unroll
  for (int e = 0; e < 16; ++e)
    b[e] = f2bf(W[(size_t)(kbase + kb + e + ((e >= 8) ? 8 : 0)) * ldw + c]);
  return b;
}
DEV void store_d(float* C, int ldc, int row0, int col0, v8f d, int lane) {
  int c = col0 + (lane & 15);
  int mb = (lane >= 16) ? 8 : 0;
#pragma unroll
  for (int e = 0; e < 8; ++e) C[(size_t)(row0 + mb + e) * ldc + c] = d[e];
}

// ---------------------------------------------------------------------------
// Kernel 1: xp = x @ W_proj (M=32768,K=256,N=1024). Tiles staged by TDM.
// ---------------------------------------------------------------------------
__global__ __launch_bounds__(256) void k_proj(const float* __restrict__ x,
                                              const float* __restrict__ Wp,
                                              float* __restrict__ xp) {
  __shared__ float Asf[128][32];
  __shared__ float Bsf[32][64];
  int tid = threadIdx.x, lane = tid & 31, wave = tid >> 5;
  int n0 = blockIdx.x * 64;
  int m0 = blockIdx.y * 128;
  v8f zero = {};
  v8f acc[4] = {zero, zero, zero, zero};
  for (int kc = 0; kc < 8; ++kc) {
#ifdef USE_TDM
    if (wave == 0) {
      // A: 128 rows x 32 cols from x (row stride 256)
      tdm_load_2d(x + (size_t)m0 * 256 + kc * 32, lds_addr_of(&Asf[0][0]), 256,
                  32768, 256, 32, 128);
      // B: 32 rows x 64 cols from W_proj (row stride 1024)
      tdm_load_2d(Wp + (size_t)(kc * 32) * 1024 + n0, lds_addr_of(&Bsf[0][0]),
                  1024, 256, 1024, 64, 32);
    }
#else
    {
      int ar = tid >> 1, ac0 = (tid & 1) * 16;
      const float* xs = x + (size_t)(m0 + ar) * 256 + kc * 32 + ac0;
#pragma unroll
      for (int s = 0; s < 4; ++s) cp16(xs + s * 4, &Asf[ar][ac0 + s * 4]);
#pragma unroll
      for (int s = 0; s < 2; ++s) {
        int lin = tid * 2 + s;
        int r = lin >> 4, c4 = (lin & 15) * 4;
        cp16(Wp + (size_t)(kc * 32 + r) * 1024 + n0 + c4, &Bsf[r][c4]);
      }
      cp_wait();
    }
#endif
    __syncthreads();
    v16bf a = load_a_ldsf(&Asf[0][0], 32, wave * 16, 0, lane);
#pragma unroll
    for (int j = 0; j < 4; ++j) {
      v16bf bm = load_b_ldsf(&Bsf[0][0], 64, j * 16, lane);
      acc[j] = wmma_bf16(a, bm, acc[j]);
    }
    __syncthreads();
  }
#pragma unroll
  for (int j = 0; j < 4; ++j)
    store_d(xp, 1024, m0 + wave * 16, n0 + j * 16, acc[j], lane);
}

// ---------------------------------------------------------------------------
// Kernel 2: masked conv as 5 shifted GEMMs per (b,g); async-LDS staging with
// per-row validity mask (not a rectangular TDM tile).
// ---------------------------------------------------------------------------
__global__ __launch_bounds__(256) void k_conv(const float* __restrict__ xp,
                                              const int* __restrict__ positions,
                                              const float* __restrict__ kern,
                                              float* __restrict__ zc) {
  __shared__ float Asf[128][32];
  __shared__ float Bsf[32][64];
  int tid = threadIdx.x, lane = tid & 31, wave = tid >> 5;
  int n0 = blockIdx.x * 64;
  int m0 = blockIdx.y * 128;
  int bz = blockIdx.z;
  int b = bz >> 2, g = bz & 3;
  v8f zero = {};
  v8f acc[4] = {zero, zero, zero, zero};
  int ar = tid >> 1, ac0 = (tid & 1) * 16;
  int tr = m0 + ar;
  int mypos = positions[b * 8192 + tr];
  for (int k = 0; k < 5; ++k) {
    int sl = tr + k - 2;
    bool valid = (sl >= 0) && (sl < 8192) &&
                 (positions[b * 8192 + sl] == mypos + (k - 2));
    const float* xs =
        xp + ((size_t)(b * 8192 + (valid ? sl : 0)) * 1024 + g * 256);
    for (int kc = 0; kc < 8; ++kc) {
      if (valid) {
#pragma unroll
        for (int s = 0; s < 4; ++s)
          cp16(xs + kc * 32 + ac0 + s * 4, &Asf[ar][ac0 + s * 4]);
      } else {
#pragma unroll
        for (int e = 0; e < 16; ++e) Asf[ar][ac0 + e] = 0.0f;
      }
      // kernel weights: stride-5 gather (cannot DMA), per-element
#pragma unroll
      for (int e = 0; e < 8; ++e) {
        int lin = tid * 8 + e;
        int r = lin >> 6, c = lin & 63;
        Bsf[r][c] =
            kern[((size_t)(g * 256 + n0 + c) * 256 + kc * 32 + r) * 5 + k];
      }
      cp_wait();
      __syncthreads();
      v16bf a = load_a_ldsf(&Asf[0][0], 32, wave * 16, 0, lane);
#pragma unroll
      for (int j = 0; j < 4; ++j) {
        v16bf bm = load_b_ldsf(&Bsf[0][0], 64, j * 16, lane);
        acc[j] = wmma_bf16(a, bm, acc[j]);
      }
      __syncthreads();
    }
  }
  float* zp = zc + (size_t)b * 8192 * 1024 + g * 256;
#pragma unroll
  for (int j = 0; j < 4; ++j)
    store_d(zp, 1024, m0 + wave * 16, n0 + j * 16, acc[j], lane);
}

// ---------------------------------------------------------------------------
// Kernel 3: hh = FFN(RoPE(x)) fused (16 tokens per block, 8 waves).
// ---------------------------------------------------------------------------
__global__ __launch_bounds__(256) void k_rope_ffn(
    const float* __restrict__ x, const int* __restrict__ positions,
    const float* __restrict__ W1, const float* __restrict__ b1,
    const float* __restrict__ ln_g, const float* __restrict__ ln_b,
    const float* __restrict__ W2, const float* __restrict__ b2,
    const float* __restrict__ ff_scale, float* __restrict__ hh) {
  __shared__ __bf16 A1[16][256];
  __shared__ float hbuf[16][256];
  __shared__ __bf16 A2[16][256];
  __shared__ float mu_s[16], rs_s[16];
  int tid = threadIdx.x, lane = tid & 31, wave = tid >> 5;
  int m0 = blockIdx.x * 16;

  {  // RoPE(x) tile -> A1 (bf16)
    int r = tid >> 4;
    int c0 = (tid & 15) * 16;
    int t = m0 + r;
    float pos = (float)positions[t];
#pragma unroll
    for (int e = 0; e < 16; e += 2) {
      int c = c0 + e;
      int p = c >> 1;
      float theta = __powf(10000.0f, -((float)(2 * p)) * (1.0f / 256.0f));
      float ang = pos * theta;
      float cs = __cosf(ang), sn = __sinf(ang);
      float x1 = x[(size_t)t * 256 + c];
      float x2 = x[(size_t)t * 256 + c + 1];
      A1[r][c] = f2bf(x1 * cs - x2 * sn);
      A1[r][c + 1] = f2bf(x1 * sn + x2 * cs);
    }
  }
  __syncthreads();

#pragma unroll
  for (int j = 0; j < 2; ++j) {  // GEMM1: 16 N-tiles
    int nt = wave + j * 8;
    v8f acc = {};
    for (int kc = 0; kc < 8; ++kc) {
      v16bf a = load_a_lds(&A1[0][0], 256, 0, kc * 32, lane);
      v16bf bm = load_b_gl(W1, 256, kc * 32, nt * 16, lane);
      acc = wmma_bf16(a, bm, acc);
    }
    int c = nt * 16 + (lane & 15);
    int mb = (lane >= 16) ? 8 : 0;
#pragma unroll
    for (int e = 0; e < 8; ++e) hbuf[mb + e][c] = acc[e] + b1[c];
  }
  __syncthreads();

  if (tid < 16) {  // LN stats
    float s = 0.f;
    for (int c = 0; c < 256; ++c) s += hbuf[tid][c];
    float mu = s * (1.0f / 256.0f);
    float vs = 0.f;
    for (int c = 0; c < 256; ++c) {
      float dd = hbuf[tid][c] - mu;
      vs += dd * dd;
    }
    mu_s[tid] = mu;
    rs_s[tid] = rsqrtf(vs * (1.0f / 256.0f) + 1e-5f);
  }
  __syncthreads();
  {  // LN + exact GELU -> A2
    int r = tid >> 4;
    int c0 = (tid & 15) * 16;
    float mu = mu_s[r], rs = rs_s[r];
#pragma unroll
    for (int e = 0; e < 16; ++e) {
      int c = c0 + e;
      float hv = (hbuf[r][c] - mu) * rs * ln_g[c] + ln_b[c];
      float ge = 0.5f * hv * (1.0f + erff(hv * 0.70710678118f));
      A2[r][c] = f2bf(ge);
    }
  }
  __syncthreads();

  for (int j = 0; j < 6; ++j) {  // GEMM2: 48 N-tiles
    int nt = wave + j * 8;
    v8f acc = {};
    for (int kc = 0; kc < 8; ++kc) {
      v16bf a = load_a_lds(&A2[0][0], 256, 0, kc * 32, lane);
      v16bf bm = load_b_gl(W2, 768, kc * 32, nt * 16, lane);
      acc = wmma_bf16(a, bm, acc);
    }
    int c = nt * 16 + (lane & 15);
    int mb = (lane >= 16) ? 8 : 0;
#pragma unroll
    for (int e = 0; e < 8; ++e) {
      int t = m0 + mb + e;
      hh[(size_t)t * 768 + c] = (acc[e] + b2[c]) * ff_scale[c];
    }
  }
}

// ---------------------------------------------------------------------------
// Kernel 4: denom[b,i,l] = sum_d |hh[b,l,i*256+d]|
// ---------------------------------------------------------------------------
__global__ __launch_bounds__(256) void k_denom(const float* __restrict__ hh,
                                               float* __restrict__ denom) {
  int idx = blockIdx.x * 256 + threadIdx.x;
  if (idx >= 4 * 3 * 8192) return;
  int l = idx & 8191;
  int bi = idx >> 13;
  int b = bi / 3, i = bi - b * 3;
  const float* p = hh + (size_t)(b * 8192 + l) * 768 + i * 256;
  float s = 0.f;
  for (int d = 0; d < 256; ++d) s += fabsf(p[d]);
  denom[idx] = s;
}

// ---------------------------------------------------------------------------
// Kernel 5: FFT long-conv recurrence, LDS-resident Stockham radix-2 (288KB of
// the 320KB WGP LDS), packed-real trick, one (b,d) sequence per workgroup.
// ---------------------------------------------------------------------------
__device__ void fft_stages(float* s0r, float* s0i, float* s1r, float* s1i,
                           float dir) {
  float *sr = s0r, *si = s0i, *dr = s1r, *di = s1i;
  int l = NFFT / 2;
  for (int st = 0; st < 14; ++st) {
    __syncthreads();
    int m = 1 << st;
    float base = dir * 3.14159265358979f / (float)l;
    for (int idx = threadIdx.x; idx < NFFT / 2; idx += 256) {
      int p = idx >> st;
      int q = idx & (m - 1);
      float ang = base * (float)p;
      float wr = __cosf(ang), wi = __sinf(ang);
      int i0 = q + m * p, i1 = q + m * (p + l);
      float arv = sr[i0], aiv = si[i0];
      float brv = sr[i1], biv = si[i1];
      int o0 = q + 2 * m * p, o1 = o0 + m;
      dr[o0] = arv + brv;
      di[o0] = aiv + biv;
      float xr = arv - brv, xi = aiv - biv;
      dr[o1] = xr * wr - xi * wi;
      di[o1] = xr * wi + xi * wr;
    }
    l >>= 1;
    float* t;
    t = sr; sr = dr; dr = t;
    t = si; si = di; di = t;
  }
  __syncthreads();  // 14 (even) stages: result back in (s0r, s0i)
}

__global__ __launch_bounds__(256) void k_fftconv(
    const float* __restrict__ zc, const float* __restrict__ hh,
    const float* __restrict__ denom, float* __restrict__ vout) {
  extern __shared__ float smem[];
  float* b0r = smem;
  float* b0i = b0r + NFFT;
  float* b1r = b0i + NFFT;
  float* b1i = b1r + NFFT;
  float* vb = b1i + NFFT;
  int tid = threadIdx.x;
  int bd = blockIdx.x;
  int b = bd >> 8, d = bd & 255;
  const float* zbase = zc + (size_t)(b * 256 + d) * (4 * 8192);
  for (int l = tid; l < 8192; l += 256) vb[l] = zbase[3 * 8192 + l];
  const float inv_n2 = 1.0f / ((float)NFFT * (float)NFFT);

  for (int i = 0; i < 3; ++i) {
    __syncthreads();
    for (int l = tid; l < NFFT; l += 256) {
      if (l < 8192) {
        float hv = hh[(size_t)(b * 8192 + l) * 768 + i * 256 + d];
        float dn = denom[(b * 3 + i) * 8192 + l] + 1e-8f;
        b0r[l] = vb[l];
        b0i[l] = hv / dn;
      } else {
        b0r[l] = 0.f;
        b0i[l] = 0.f;
      }
    }
    fft_stages(b0r, b0i, b1r, b1i, -1.0f);
    __syncthreads();
    for (int k = tid; k < NFFT; k += 256) {
      int nk = (NFFT - k) & (NFFT - 1);
      float x1 = b0r[k], y1 = b0i[k];
      float x2 = b0r[nk], y2 = b0i[nk];
      float vr = 0.5f * (x1 + x2), vi = 0.5f * (y1 - y2);
      float hr = 0.5f * (y1 + y2), hi = -0.5f * (x1 - x2);
      b1r[k] = vr * hr - vi * hi;
      b1i[k] = vr * hi + vi * hr;
    }
    fft_stages(b1r, b1i, b0r, b0i, 1.0f);
    __syncthreads();
    for (int l = tid; l < 8192; l += 256)
      vb[l] = zbase[(size_t)i * 8192 + l] * (b1r[l] * inv_n2);
  }
  __syncthreads();
  for (int l = tid; l < 8192; l += 256)
    vout[(size_t)(b * 256 + d) * 8192 + l] = vb[l];
}

// ---------------------------------------------------------------------------
// Kernel 6: out = v^T @ W_out + b_out; A (channel-major) + B staged by TDM.
// ---------------------------------------------------------------------------
__global__ __launch_bounds__(256) void k_out(const float* __restrict__ vout,
                                             const float* __restrict__ Wo,
                                             const float* __restrict__ bo,
                                             float* __restrict__ out) {
  __shared__ float Asf_t[32][128];  // [k][token]
  __shared__ float Bsf[32][64];
  int tid = threadIdx.x, lane = tid & 31, wave = tid >> 5;
  int n0 = blockIdx.x * 64;
  int m0 = blockIdx.y * 128;
  int b = m0 >> 13, l0 = m0 & 8191;
  v8f zero = {};
  v8f acc[4] = {zero, zero, zero, zero};
  for (int kc = 0; kc < 8; ++kc) {
#ifdef USE_TDM
    if (wave == 0) {
      // A: 32 channel-rows x 128 tokens from (B,D,L) (row stride 8192)
      tdm_load_2d(vout + (size_t)(b * 256 + kc * 32) * 8192 + l0,
                  lds_addr_of(&Asf_t[0][0]), 8192, 1024, 8192, 128, 32);
      // B: 32 rows x 64 cols from W_out (row stride 256)
      tdm_load_2d(Wo + (size_t)(kc * 32) * 256 + n0, lds_addr_of(&Bsf[0][0]),
                  256, 256, 256, 64, 32);
    }
#else
    {
      int kk = tid >> 3;
      int ts = (tid & 7) * 16;
      const float* src =
          vout + (size_t)(b * 256 + kc * 32 + kk) * 8192 + l0 + ts;
#pragma unroll
      for (int s = 0; s < 4; ++s) cp16(src + s * 4, &Asf_t[kk][ts + s * 4]);
#pragma unroll
      for (int s = 0; s < 2; ++s) {
        int lin = tid * 2 + s;
        int r = lin >> 4, c4 = (lin & 15) * 4;
        cp16(Wo + (size_t)(kc * 32 + r) * 256 + n0 + c4, &Bsf[r][c4]);
      }
      cp_wait();
    }
#endif
    __syncthreads();
    v16bf a = load_a_ldsf_t(&Asf_t[0][0], 128, wave * 16, 0, lane);
#pragma unroll
    for (int j = 0; j < 4; ++j) {
      v16bf bm = load_b_ldsf(&Bsf[0][0], 64, j * 16, lane);
      acc[j] = wmma_bf16(a, bm, acc[j]);
    }
    __syncthreads();
  }
#pragma unroll
  for (int j = 0; j < 4; ++j) {
    int c = n0 + j * 16 + (lane & 15);
    int mb = (lane >= 16) ? 8 : 0;
#pragma unroll
    for (int e = 0; e < 8; ++e)
      out[(size_t)(m0 + wave * 16 + mb + e) * 256 + c] = acc[j][e] + bo[c];
  }
}

// ---------------------------------------------------------------------------
extern "C" void kernel_launch(void* const* d_in, const int* in_sizes, int n_in,
                              void* d_out, int out_size, void* d_ws,
                              size_t ws_size, hipStream_t stream) {
  (void)in_sizes; (void)n_in; (void)out_size; (void)ws_size;
  const float* x = (const float*)d_in[0];
  const int* pos = (const int*)d_in[1];
  const float* Wp = (const float*)d_in[2];
  const float* kern = (const float*)d_in[3];
  const float* W1 = (const float*)d_in[4];
  const float* b1 = (const float*)d_in[5];
  const float* lng = (const float*)d_in[6];
  const float* lnb = (const float*)d_in[7];
  const float* W2 = (const float*)d_in[8];
  const float* b2 = (const float*)d_in[9];
  const float* ffs = (const float*)d_in[10];
  const float* Wo = (const float*)d_in[11];
  const float* bo = (const float*)d_in[12];
  float* out = (float*)d_out;
  float* ws = (float*)d_ws;

  float* xp = ws;                // 33,554,432 f32 (B,L,1024)
  float* zc = xp + 33554432;     // 33,554,432 f32 (B,L,4,256)
  float* hh = zc + 33554432;     // 25,165,824 f32 (B,L,768)
  float* denom = hh + 25165824;  //     98,304 f32 (B,3,L)
  float* vout = xp;              // reuse xp region for (B,D,L)

  k_proj<<<dim3(16, 256), 256, 0, stream>>>(x, Wp, xp);
  k_conv<<<dim3(4, 64, 16), 256, 0, stream>>>(xp, pos, kern, zc);
  k_rope_ffn<<<dim3(2048), 256, 0, stream>>>(x, pos, W1, b1, lng, lnb, W2, b2,
                                             ffs, hh);
  k_denom<<<dim3(384), 256, 0, stream>>>(hh, denom);
  k_fftconv<<<dim3(1024), 256, 294912, stream>>>(zc, hh, denom, vout);
  k_out<<<dim3(4, 256), 256, 0, stream>>>(vout, Wo, bo, out);
}
